// MoEWithDeepEP_76441827935054
// MI455X (gfx1250) — matrix operand order repaired and checked by reading, not compiled, not run on hardware
//
#include <hip/hip_runtime.h>
#include <hip/hip_bf16.h>
#include <stdint.h>

// ---------------- types ----------------
typedef __bf16 bf16;
typedef bf16  v16bf __attribute__((ext_vector_type(16)));
typedef bf16  v8bf  __attribute__((ext_vector_type(8)));
typedef float v8f   __attribute__((ext_vector_type(8)));

// ---------------- config ----------------
#define NE      64      // experts
#define TOPK    2
#define DM      512     // model dim
#define HM      256     // per-expert hidden
#define HS      512     // shared hidden (H*NSH)
#define CAPE    1024    // per-expert capacity
#define NTOK    8192    // 4*2048 tokens
#define RSCALE  2.5f

// ---------------- helpers ----------------
__device__ __forceinline__ bf16 f2bf(float f) {
    unsigned u = __builtin_bit_cast(unsigned, f);
    unsigned r = u + 0x7FFFu + ((u >> 16) & 1u);       // RNE
    unsigned short h = (unsigned short)(r >> 16);
    return __builtin_bit_cast(bf16, h);
}

__device__ __forceinline__ v8f wmma_bf16(v16bf a, v16bf b, v8f c) {
    // (neg_a, A, neg_b, B, c_mod, C, reuse_a, reuse_b)
    return __builtin_amdgcn_wmma_f32_16x16x32_bf16(false, a, false, b, (short)0, c,
                                                   false, false);
}

// Async copy 16 bytes global -> LDS (gfx1250 GLOBAL_LOAD_ASYNC_TO_LDS_B128).
// LDS byte offset = low 32 bits of the generic pointer (ISA 10.2: LDS aperture
// addresses carry the LDS offset in addr[31:0]).
__device__ __forceinline__ void async_copy_b128(void* lds_dst, const void* gsrc) {
    unsigned lds_off = (unsigned)(uintptr_t)lds_dst;
    asm volatile("global_load_async_to_lds_b128 %0, %1, off"
                 :: "v"(lds_off), "v"(gsrc)
                 : "memory");
}
__device__ __forceinline__ void wait_async0() {
    asm volatile("s_wait_asynccnt 0x0" ::: "memory");
}

// A-fragment: 16x32 bf16 tile, row-major in LDS with row stride `ld` elements.
// Lane l: m = l&15, half = l>>4. Elements 0..15 = K(16h..16h+15).
__device__ __forceinline__ v16bf load_a_frag(const bf16* row, int half) {
    v8bf lo = *(const v8bf*)(row + half * 16);
    v8bf hi = *(const v8bf*)(row + half * 16 + 8);
    v16bf a;
#pragma unroll
    for (int i = 0; i < 8; ++i) { a[i] = lo[i]; a[8 + i] = hi[i]; }
    return a;
}

// ---------------- kernel: zero expert counters ----------------
__global__ void zero_counts_kernel(int* counts) {
    if (threadIdx.x < NE) counts[threadIdx.x] = 0;
}

// ---------------- kernel: fp32 -> bf16 copy ----------------
__global__ void cvt_bf16_kernel(const float* __restrict__ src, bf16* __restrict__ dst, int total) {
    for (int i = blockIdx.x * blockDim.x + threadIdx.x; i < total; i += gridDim.x * blockDim.x)
        dst[i] = f2bf(src[i]);
}

// ---------------- kernel: pack weights into WMMA B layout ----------------
// src: [slices][Kt][N] row-major fp32.  dst: [slices][Kt/32][N][32] bf16.
// B-fragment for (kc, n, half) is then 16 contiguous bf16 at dst + kc*N*32 + n*32 + half*16.
__global__ void pack_b_kernel(const float* __restrict__ src, bf16* __restrict__ dst,
                              int Kt, int N, int total) {
    int kn = Kt * N;
    for (int i = blockIdx.x * blockDim.x + threadIdx.x; i < total; i += gridDim.x * blockDim.x) {
        int sl = i / kn;
        int r  = i - sl * kn;
        int k  = r / N;
        int n  = r - k * N;
        dst[(size_t)sl * kn + (size_t)(k >> 5) * (N * 32) + (n << 5) + (k & 31)] = f2bf(src[i]);
    }
}

// ---------------- kernel: router (sigmoid top-2) + dispatch ----------------
__global__ __launch_bounds__(64) void router_kernel(
    const float* __restrict__ x, const float* __restrict__ gate_w,
    int* __restrict__ counts, int* __restrict__ expert_rows, float* __restrict__ score_es)
{
    __shared__ float xr[DM];
    __shared__ float sc[NE];
    const int tok = blockIdx.x;
    const int tid = threadIdx.x;

    // stage token row
#pragma unroll
    for (int i = 0; i < DM / 64; ++i) xr[tid + i * 64] = x[(size_t)tok * DM + tid + i * 64];
    __syncthreads();

    // one expert per thread
    const float* g = gate_w + (size_t)tid * DM;
    float acc = 0.f;
#pragma unroll 8
    for (int d = 0; d < DM; ++d) acc += xr[d] * g[d];
    sc[tid] = 1.f / (1.f + __expf(-acc));
    __syncthreads();

    if (tid == 0) {
        float s1 = -1e30f, s2 = -1e30f; int i1 = 0, i2 = 0;
        for (int e2 = 0; e2 < NE; ++e2) {
            float v = sc[e2];
            if (v > s1) { s2 = s1; i2 = i1; s1 = v; i1 = e2; }
            else if (v > s2) { s2 = v; i2 = e2; }
        }
        float denom = s1 + s2 + 1e-20f;
        float r1 = s1 / denom * RSCALE;
        float r2 = s2 / denom * RSCALE;
        int slot1 = atomicAdd(&counts[i1], 1);
        if (slot1 < CAPE) { expert_rows[i1 * CAPE + slot1] = tok; score_es[i1 * CAPE + slot1] = r1; }
        int slot2 = atomicAdd(&counts[i2], 1);
        if (slot2 < CAPE) { expert_rows[i2 * CAPE + slot2] = tok; score_es[i2 * CAPE + slot2] = r2; }
    }
}

// ---------------- kernel: shared-expert fused SwiGLU FFN ----------------
// grid = NTOK/32 blocks, 512 threads (16 waves). Writes (initializes) d_out.
__global__ __launch_bounds__(512) void shared_ffn_kernel(
    const bf16* __restrict__ Xb, const bf16* __restrict__ sw1p,
    const bf16* __restrict__ sw3p, const bf16* __restrict__ sw2p,
    float* __restrict__ out)
{
    constexpr int LD = 520;                       // row stride (pad 8) for X and H tiles
    __shared__ __align__(16) bf16 smem[32 * LD];  // X tile, then reused as H tile

    const int tid  = threadIdx.x;
    const int lane = tid & 31, wave = tid >> 5;
    const int half = lane >> 4, l15 = lane & 15;
    const int mrow0 = blockIdx.x * 32;

    // stage X tile 32 x 512 bf16 via async global->LDS DMA
    {
        int r = tid >> 4;
        int cseg = (tid & 15) * 32;
        const bf16* src = Xb + (size_t)(mrow0 + r) * DM + cseg;
        bf16* dst = &smem[r * LD + cseg];
#pragma unroll
        for (int c = 0; c < 32; c += 8)
            async_copy_b128(dst + c, src + c);
    }
    wait_async0();
    __syncthreads();

    // ---- stage 1: X@sw1, X@sw3 ; wave handles hidden cols [wave*32, wave*32+32) ----
    const int wnb = wave * 32;
    v8f acc1[2][2], acc3[2][2];
#pragma unroll
    for (int a = 0; a < 2; ++a)
#pragma unroll
        for (int b = 0; b < 2; ++b) {
            acc1[a][b] = (v8f){0, 0, 0, 0, 0, 0, 0, 0};
            acc3[a][b] = (v8f){0, 0, 0, 0, 0, 0, 0, 0};
        }

    for (int kc = 0; kc < DM / 32; ++kc) {
        v16bf a0 = load_a_frag(&smem[(l15) * LD + kc * 32], half);
        v16bf a1 = load_a_frag(&smem[(16 + l15) * LD + kc * 32], half);
        if (kc + 1 < DM / 32)
            __builtin_prefetch(sw1p + (size_t)(kc + 1) * (HS * 32) + (wnb + l15) * 32, 0, 0);
#pragma unroll
        for (int nt = 0; nt < 2; ++nt) {
            int n = wnb + nt * 16 + l15;
            v16bf b1 = *(const v16bf*)(sw1p + (size_t)kc * (HS * 32) + n * 32 + half * 16);
            v16bf b3 = *(const v16bf*)(sw3p + (size_t)kc * (HS * 32) + n * 32 + half * 16);
            acc1[0][nt] = wmma_bf16(a0, b1, acc1[0][nt]);
            acc1[1][nt] = wmma_bf16(a1, b1, acc1[1][nt]);
            acc3[0][nt] = wmma_bf16(a0, b3, acc3[0][nt]);
            acc3[1][nt] = wmma_bf16(a1, b3, acc3[1][nt]);
        }
    }
    __syncthreads();   // all waves done reading X tile

    // silu(g)*u -> H tile (bf16) in the same LDS
#pragma unroll
    for (int mt2 = 0; mt2 < 2; ++mt2)
#pragma unroll
        for (int nt = 0; nt < 2; ++nt)
#pragma unroll
            for (int i = 0; i < 8; ++i) {
                float g = acc1[mt2][nt][i];
                float hv = (g / (1.f + __expf(-g))) * acc3[mt2][nt][i];
                int m = mt2 * 16 + half * 8 + i;
                int n = wnb + nt * 16 + l15;
                smem[m * LD + n] = f2bf(hv);
            }
    __syncthreads();

    // ---- stage 2: H@sw2 ; wave handles out cols [wave*32, wave*32+32) ----
    v8f accY[2][2];
#pragma unroll
    for (int a = 0; a < 2; ++a)
#pragma unroll
        for (int b = 0; b < 2; ++b) accY[a][b] = (v8f){0, 0, 0, 0, 0, 0, 0, 0};

    for (int kc = 0; kc < HS / 32; ++kc) {
        v16bf a0 = load_a_frag(&smem[(l15) * LD + kc * 32], half);
        v16bf a1 = load_a_frag(&smem[(16 + l15) * LD + kc * 32], half);
#pragma unroll
        for (int nt = 0; nt < 2; ++nt) {
            int n = wnb + nt * 16 + l15;
            v16bf b = *(const v16bf*)(sw2p + (size_t)kc * (DM * 32) + n * 32 + half * 16);
            accY[0][nt] = wmma_bf16(a0, b, accY[0][nt]);
            accY[1][nt] = wmma_bf16(a1, b, accY[1][nt]);
        }
    }

    // store (initializes d_out)
#pragma unroll
    for (int mt2 = 0; mt2 < 2; ++mt2)
#pragma unroll
        for (int i = 0; i < 8; ++i) {
            int m = mt2 * 16 + half * 8 + i;
#pragma unroll
            for (int nt = 0; nt < 2; ++nt) {
                int n = wnb + nt * 16 + l15;
                out[(size_t)(mrow0 + m) * DM + n] = accY[mt2][nt][i];
            }
        }
}

// ---------------- kernel: routed-expert fused SwiGLU FFN ----------------
// grid = NE * (CAPE/32) blocks, 256 threads (8 waves). atomicAdd into d_out.
__global__ __launch_bounds__(256) void expert_ffn_kernel(
    const bf16* __restrict__ Xb, const bf16* __restrict__ w1p,
    const bf16* __restrict__ w3p, const bf16* __restrict__ w2p,
    const int* __restrict__ counts, const int* __restrict__ expert_rows,
    const float* __restrict__ score_es, float* __restrict__ out)
{
    constexpr int MT  = CAPE / 32;
    constexpr int LDX = 520;   // X tile row stride
    constexpr int LDH = 264;   // H tile row stride
    const int e  = blockIdx.x / MT;
    const int mt = blockIdx.x % MT;
    int rows = counts[e];
    rows = rows < CAPE ? rows : CAPE;
    if (mt * 32 >= rows) return;          // block-uniform: EXEC stays all-1s

    __shared__ __align__(16) bf16 smem[32 * LDX];   // X tile, reused as H tile
    __shared__ int   rid_s[32];
    __shared__ float sc_s[32];

    const int tid  = threadIdx.x;
    const int lane = tid & 31, wave = tid >> 5;
    const int half = lane >> 4, l15 = lane & 15;

    // gather-stage X tile: 32 rows (expert slots) x 512 cols bf16.
    // Valid rows: async global->LDS DMA (EXEC-masked per lane). Pad rows: zero-fill.
    {
        int r = tid >> 3;
        int cseg = (tid & 7) * 64;
        int gslot = mt * 32 + r;
        int rid = (gslot < rows) ? expert_rows[e * CAPE + gslot] : -1;
        if (cseg == 0) {
            rid_s[r] = rid;
            sc_s[r]  = (gslot < rows) ? score_es[e * CAPE + gslot] : 0.f;
        }
        if (rid >= 0) {
            const bf16* src = Xb + (size_t)rid * DM + cseg;
            bf16* dst = &smem[r * LDX + cseg];
#pragma unroll
            for (int c = 0; c < 64; c += 8)
                async_copy_b128(dst + c, src + c);
        } else {
            v8bf z;
#pragma unroll
            for (int i = 0; i < 8; ++i) z[i] = __builtin_bit_cast(bf16, (unsigned short)0);
#pragma unroll
            for (int c = 0; c < 64; c += 8)
                *(v8bf*)(&smem[r * LDX + cseg + c]) = z;
        }
    }
    wait_async0();
    __syncthreads();

    // ---- stage 1: X@w1[e], X@w3[e] ; wave handles hidden cols [wave*32, +32) ----
    const bf16* w1e = w1p + (size_t)e * DM * HM;
    const bf16* w3e = w3p + (size_t)e * DM * HM;
    const int wnb = wave * 32;
    v8f acc1[2][2], acc3[2][2];
#pragma unroll
    for (int a = 0; a < 2; ++a)
#pragma unroll
        for (int b = 0; b < 2; ++b) {
            acc1[a][b] = (v8f){0, 0, 0, 0, 0, 0, 0, 0};
            acc3[a][b] = (v8f){0, 0, 0, 0, 0, 0, 0, 0};
        }

    for (int kc = 0; kc < DM / 32; ++kc) {
        v16bf a0 = load_a_frag(&smem[(l15) * LDX + kc * 32], half);
        v16bf a1 = load_a_frag(&smem[(16 + l15) * LDX + kc * 32], half);
        if (kc + 1 < DM / 32)
            __builtin_prefetch(w1e + (size_t)(kc + 1) * (HM * 32) + (wnb + l15) * 32, 0, 0);
#pragma unroll
        for (int nt = 0; nt < 2; ++nt) {
            int n = wnb + nt * 16 + l15;
            v16bf b1 = *(const v16bf*)(w1e + (size_t)kc * (HM * 32) + n * 32 + half * 16);
            v16bf b3 = *(const v16bf*)(w3e + (size_t)kc * (HM * 32) + n * 32 + half * 16);
            acc1[0][nt] = wmma_bf16(a0, b1, acc1[0][nt]);
            acc1[1][nt] = wmma_bf16(a1, b1, acc1[1][nt]);
            acc3[0][nt] = wmma_bf16(a0, b3, acc3[0][nt]);
            acc3[1][nt] = wmma_bf16(a1, b3, acc3[1][nt]);
        }
    }
    __syncthreads();   // all waves done reading X tile

    // silu(g)*u -> H tile bf16 (32 x 256, stride LDH) in same LDS
#pragma unroll
    for (int mt2 = 0; mt2 < 2; ++mt2)
#pragma unroll
        for (int nt = 0; nt < 2; ++nt)
#pragma unroll
            for (int i = 0; i < 8; ++i) {
                float g = acc1[mt2][nt][i];
                float hv = (g / (1.f + __expf(-g))) * acc3[mt2][nt][i];
                int m = mt2 * 16 + half * 8 + i;
                int n = wnb + nt * 16 + l15;
                smem[m * LDH + n] = f2bf(hv);
            }
    __syncthreads();

    // ---- stage 2: H@w2[e] ; wave handles out cols [wave*64, +64) ----
    const bf16* w2e = w2p + (size_t)e * HM * DM;
    const int wob = wave * 64;
    v8f accY[2][4];
#pragma unroll
    for (int a = 0; a < 2; ++a)
#pragma unroll
        for (int b = 0; b < 4; ++b) accY[a][b] = (v8f){0, 0, 0, 0, 0, 0, 0, 0};

    for (int kc = 0; kc < HM / 32; ++kc) {
        v16bf a0 = load_a_frag(&smem[(l15) * LDH + kc * 32], half);
        v16bf a1 = load_a_frag(&smem[(16 + l15) * LDH + kc * 32], half);
#pragma unroll
        for (int nt = 0; nt < 4; ++nt) {
            int n = wob + nt * 16 + l15;
            v16bf b = *(const v16bf*)(w2e + (size_t)kc * (DM * 32) + n * 32 + half * 16);
            accY[0][nt] = wmma_bf16(a0, b, accY[0][nt]);
            accY[1][nt] = wmma_bf16(a1, b, accY[1][nt]);
        }
    }

    // epilogue: scale by routing score, accumulate into d_out
#pragma unroll
    for (int mt2 = 0; mt2 < 2; ++mt2)
#pragma unroll
        for (int i = 0; i < 8; ++i) {
            int m = mt2 * 16 + half * 8 + i;
            if (mt * 32 + m < rows) {
                int rid = rid_s[m];
                float s = sc_s[m];
#pragma unroll
                for (int nt = 0; nt < 4; ++nt) {
                    int n = wob + nt * 16 + l15;
                    atomicAdd(&out[(size_t)rid * DM + n], accY[mt2][nt][i] * s);
                }
            }
        }
}

// ---------------- host launcher ----------------
extern "C" void kernel_launch(void* const* d_in, const int* in_sizes, int n_in,
                              void* d_out, int out_size, void* d_ws, size_t ws_size,
                              hipStream_t stream) {
    const float* x      = (const float*)d_in[0];
    const float* gate_w = (const float*)d_in[1];
    const float* w1     = (const float*)d_in[2];
    const float* w3     = (const float*)d_in[3];
    const float* w2     = (const float*)d_in[4];
    const float* sw1    = (const float*)d_in[5];
    const float* sw3    = (const float*)d_in[6];
    const float* sw2    = (const float*)d_in[7];
    float* out = (float*)d_out;

    // workspace carve-out (256B aligned)
    char* ws = (char*)d_ws;
    size_t off = 0;
    auto carve = [&](size_t bytes) -> char* {
        char* p = ws + off;
        off = (off + bytes + 255) & ~(size_t)255;
        return p;
    };
    bf16* w1p  = (bf16*)carve((size_t)NE * DM * HM * 2);
    bf16* w3p  = (bf16*)carve((size_t)NE * DM * HM * 2);
    bf16* w2p  = (bf16*)carve((size_t)NE * HM * DM * 2);
    bf16* sw1p = (bf16*)carve((size_t)DM * HS * 2);
    bf16* sw3p = (bf16*)carve((size_t)DM * HS * 2);
    bf16* sw2p = (bf16*)carve((size_t)HS * DM * 2);
    bf16* Xb   = (bf16*)carve((size_t)NTOK * DM * 2);
    int*  counts      = (int*)carve(NE * sizeof(int));
    int*  expert_rows = (int*)carve((size_t)NE * CAPE * sizeof(int));
    float* score_es   = (float*)carve((size_t)NE * CAPE * sizeof(float));
    (void)ws_size; (void)in_sizes; (void)n_in; (void)out_size;

    // 1) zero per-expert counters
    zero_counts_kernel<<<1, 64, 0, stream>>>(counts);

    // 2) pack weights to bf16 WMMA-B layout
    pack_b_kernel<<<4096, 256, 0, stream>>>(w1,  w1p,  DM, HM, NE * DM * HM);
    pack_b_kernel<<<4096, 256, 0, stream>>>(w3,  w3p,  DM, HM, NE * DM * HM);
    pack_b_kernel<<<4096, 256, 0, stream>>>(w2,  w2p,  HM, DM, NE * HM * DM);
    pack_b_kernel<<<1024, 256, 0, stream>>>(sw1, sw1p, DM, HS, DM * HS);
    pack_b_kernel<<<1024, 256, 0, stream>>>(sw3, sw3p, DM, HS, DM * HS);
    pack_b_kernel<<<1024, 256, 0, stream>>>(sw2, sw2p, HS, DM, HS * DM);

    // 3) token activations to bf16
    cvt_bf16_kernel<<<4096, 256, 0, stream>>>(x, Xb, NTOK * DM);

    // 4) router + dispatch (fp32 precision for gating)
    router_kernel<<<NTOK, 64, 0, stream>>>(x, gate_w, counts, expert_rows, score_es);

    // 5) shared expert FFN (writes/initializes d_out)
    shared_ffn_kernel<<<NTOK / 32, 512, 0, stream>>>(Xb, sw1p, sw3p, sw2p, out);

    // 6) routed experts (accumulates into d_out)
    expert_ffn_kernel<<<NE * (CAPE / 32), 256, 0, stream>>>(
        Xb, w1p, w3p, w2p, counts, expert_rows, score_es, out);
}
